// Wave_Fusion_Model_55714315764077
// MI455X (gfx1250) — matrix operand order repaired and checked by reading, not compiled, not run on hardware
//
#include <hip/hip_runtime.h>
#include <hip/hip_bf16.h>

typedef __attribute__((ext_vector_type(16))) _Float16 v16h;
typedef __attribute__((ext_vector_type(8)))  float    v8f;
typedef __attribute__((ext_vector_type(4)))  unsigned int u32x4;
typedef __attribute__((ext_vector_type(8)))  unsigned int u32x8;

#define N_LEADS 61
#define N_BATCH 128

static __device__ inline void wait_asynccnt0() {
#if __has_builtin(__builtin_amdgcn_s_wait_asynccnt)
    __builtin_amdgcn_s_wait_asynccnt(0);
#else
    asm volatile("s_wait_asynccnt 0" ::: "memory");
#endif
}

// One workgroup = one lead x 2 samples. 256 threads = 8 wave32.
__global__ __launch_bounds__(256) void lead_cnn_kernel(
    const float* __restrict__ x,
    const float* __restrict__ w1, const float* __restrict__ b1,
    const float* __restrict__ w2, const float* __restrict__ b2,
    const float* __restrict__ w3, const float* __restrict__ b3,
    const float* __restrict__ w4, const float* __restrict__ b4,
    const float* __restrict__ fcw, const float* __restrict__ fcb,
    float* __restrict__ preds)          // [B][LEADS][3]
{
    const int l    = blockIdx.y;        // lead
    const int cb   = blockIdx.x;        // 2-sample chunk
    const int tid  = threadIdx.x;
    const int lane = tid & 31;
    const int wave = tid >> 5;
    const int hi   = (lane >= 16) ? 1 : 0;
    const int col  = lane & 15;

    // ---- LDS (~63.6 KB total) ----
    __shared__ float    s_w2f[1536];              //  6144 B : conv2 weights via TDM
    __shared__ float    s_w1[96], s_b1[8], s_b2[16], s_b3[32];
    __shared__ _Float16 s_h1p[2 * 8 * 16 * 63];   // 32256 B : conv1 pooled+relu
    __shared__ _Float16 s_h2 [2 * 16 * 8 * 32];   // 16384 B : conv2 raw
    __shared__ _Float16 s_h2p[2 * 16 * 4 * 16];   //  4096 B
    __shared__ _Float16 s_h3 [2 * 32 * 4 * 8];    //  4096 B
    __shared__ _Float16 s_h3p[2 * 32 * 2 * 4];    //  1024 B
    __shared__ float    s_h4 [2 * 64];            //   512 B
    __shared__ float    s_pred[8];                //    32 B

    const float* w1g  = w1  + l * 96;
    const float* b1g  = b1  + l * 8;
    const float* w2g  = w2  + l * 1536;
    const float* b2g  = b2  + l * 16;
    const float* w3g  = w3  + l * 6144;
    const float* b3g  = b3  + l * 32;
    const float* w4g  = w4  + l * 16384;
    const float* b4g  = b4  + l * 64;
    const float* fcwg = fcw + l * 192;
    const float* fcbg = fcb + l * 3;

    // ---- Tensor Data Mover: DMA this lead's w2 (1536 dwords) into LDS, overlapped
    //      with conv1. TDM ignores EXEC and issues once per executing wave -> wave 0 only.
    if (tid < 32) {
        unsigned            lds_w2 = (unsigned)(size_t)&s_w2f[0];
        unsigned long long  ga     = (unsigned long long)(size_t)w2g;
        // D# group 0: count=1 | lds_addr | global_addr[56:0] | type=2
        u32x4 g0 = { 1u,
                     lds_w2,
                     (unsigned)(ga & 0xFFFFFFFFu),
                     ((unsigned)((ga >> 32) & 0x01FFFFFFu)) | 0x80000000u };
        // D# group 1: data_size=4B, tensor_dim0=1536, tensor_dim1=1,
        //             tile_dim0=1536, tile_dim1=1, tensor_dim0_stride=1536
        u32x8 g1 = { 0x00020000u,
                     (1536u & 0xFFFFu) << 16,
                     (1536u >> 16) | (1u << 16),
                     (1536u << 16),
                     1u,
                     1536u,
                     0u, 0u };
        asm volatile("tensor_load_to_lds %0, %1" :: "s"(g0), "s"(g1) : "memory");
    }

    // ---- stage tiny per-lead params ----
    for (int i = tid; i < 96; i += 256) s_w1[i] = w1g[i];
    if (tid < 8)  s_b1[tid] = b1g[tid];
    if (tid < 16) s_b2[tid] = b2g[tid];
    if (tid < 32) s_b3[tid] = b3g[tid];
    __syncthreads();

    // ================= conv1 (1->8, k3x4, s(1,2), pad(1,1)(2,2)) + maxpool2 + relu =================
    // output h1p: (2,8,16,63). 16128 elements / 256 threads = 63 each, no tail.
    for (int idx = tid; idx < 2 * 8 * 16 * 63; idx += 256) {
        int pw = idx % 63;
        int t  = idx / 63;
        int ph = t & 15;  t >>= 4;
        int oc = t & 7;
        int s  = t >> 3;
        const float* xg = x + ((size_t)(cb * 2 + s) * N_LEADS + l) * 8000;

        // union window of the 2x2 pooled conv taps: rows 2ph-1..2ph+2, cols 4pw-2..4pw+3
        float xw[4][6];
        for (int r = 0; r < 4; ++r) {
            int y = 2 * ph - 1 + r;
            for (int c = 0; c < 6; ++c) {
                int xx = 4 * pw - 2 + c;
                xw[r][c] = ((unsigned)y < 32u && (unsigned)xx < 250u) ? xg[y * 250 + xx] : 0.f;
            }
        }
        float m = -3.4e38f;
        for (int dh = 0; dh < 2; ++dh)
            for (int dw = 0; dw < 2; ++dw) {
                float acc = s_b1[oc];
                for (int kh = 0; kh < 3; ++kh)
                    for (int kw = 0; kw < 4; ++kw)
                        acc += s_w1[oc * 12 + kh * 4 + kw] * xw[dh + kh][2 * dw + kw];
                m = fmaxf(m, acc);
            }
        s_h1p[idx] = (_Float16)fmaxf(m, 0.f);
    }
    // TDM drain (wave 0 owns the TENSORcnt), then make w2/h1p visible to all.
    if (tid < 32) __builtin_amdgcn_s_wait_tensorcnt(0);
    __syncthreads();

    // ================= conv2 (8->16, k4x3, s(2,2), pad(1,1)) via WMMA implicit GEMM =================
    // M=16 (oc), K=96 (ic,kh,kw), N=512 positions (2 samples x 8 x 32) -> 32 N-tiles, 3 K-chunks.
    {
        // A fragments (weights, from LDS), held in registers for all N tiles of this wave.
        v16h a2[3];
        for (int kc = 0; kc < 3; ++kc) {
            v16h a;
            for (int i = 0; i < 16; ++i) {
                int kk = i + (hi ? 16 : 0);   // contiguous k = 16*hi + i, same order as B
                a[i] = (_Float16)s_w2f[col * 96 + kc * 32 + kk];
            }
            a2[kc] = a;
        }
        for (int t = 0; t < 4; ++t) {
            int nt = wave + t * 8;
            int n  = nt * 16 + col;
            int s  = n >> 8, rem = n & 255, oh = rem >> 5, ow = rem & 31;
            v8f c = {};
            for (int kc = 0; kc < 3; ++kc) {
                v16h b;
                int kbase = kc * 32 + (hi ? 16 : 0);
                for (int i = 0; i < 16; ++i) {
                    int k  = kbase + i;
                    int ic = k / 12, r = k % 12, kh = r / 3, kw = r % 3;
                    int y  = oh * 2 - 1 + kh;
                    int xx = ow * 2 - 1 + kw;
                    _Float16 v = (_Float16)0.f;
                    if ((unsigned)y < 16u && (unsigned)xx < 63u)
                        v = s_h1p[((s * 8 + ic) * 16 + y) * 63 + xx];
                    b[i] = v;
                }
                c = __builtin_amdgcn_wmma_f32_16x16x32_f16(false, a2[kc], false, b,
                                                           (short)0, c, false, false);
            }
            // D tile -> h2 (+bias), per 32-bit C/D layout
            for (int r = 0; r < 8; ++r) {
                int oc = r + (hi ? 8 : 0);
                float v = c[r] + s_b2[oc];
                s_h2[((s * 16 + oc) * 8 + oh) * 32 + ow] = (_Float16)v;
            }
        }
    }
    __syncthreads();

    // maxpool2 + relu -> h2p (2,16,4,16)
    for (int idx = tid; idx < 2048; idx += 256) {
        int pw = idx & 15; int t = idx >> 4;
        int ph = t & 3;    t >>= 2;
        int oc = t & 15;   int s = t >> 4;
        const _Float16* base = &s_h2[((s * 16 + oc) * 8 + 2 * ph) * 32 + 2 * pw];
        float m = fmaxf(fmaxf((float)base[0], (float)base[1]),
                        fmaxf((float)base[32], (float)base[33]));
        s_h2p[idx] = (_Float16)fmaxf(m, 0.f);
    }
    __syncthreads();

    // ================= conv3 (16->32, k3x4, s(1,2), pad(1,1)) via WMMA =================
    // M=32 -> 2 tiles, K=192 -> 6 chunks, N=64 (2 samples x 4 x 8) -> 4 tiles. 8 tile-pairs, 1/wave.
    {
        int mt = wave >> 2;
        int nt = wave & 3;
        int n  = nt * 16 + col;
        int s  = n >> 5, rem = n & 31, oh = rem >> 3, ow = rem & 7;
        v8f c = {};
        for (int kc = 0; kc < 6; ++kc) {
            v16h a;
            for (int i = 0; i < 16; ++i) {
                int kk = i + (hi ? 16 : 0);
                a[i] = (_Float16)w3g[(mt * 16 + col) * 192 + kc * 32 + kk];
            }
            v16h b;
            int kbase = kc * 32 + (hi ? 16 : 0);
            for (int i = 0; i < 16; ++i) {
                int k  = kbase + i;
                int ic = k / 12, r = k % 12, kh = r >> 2, kw = r & 3;
                int y  = oh - 1 + kh;
                int xx = ow * 2 - 1 + kw;
                _Float16 v = (_Float16)0.f;
                if ((unsigned)y < 4u && (unsigned)xx < 16u)
                    v = s_h2p[((s * 16 + ic) * 4 + y) * 16 + xx];
                b[i] = v;
            }
            c = __builtin_amdgcn_wmma_f32_16x16x32_f16(false, a, false, b,
                                                       (short)0, c, false, false);
        }
        for (int r = 0; r < 8; ++r) {
            int oc = mt * 16 + r + (hi ? 8 : 0);
            float v = c[r] + s_b3[oc];
            s_h3[((s * 32 + oc) * 4 + oh) * 8 + ow] = (_Float16)v;
        }
    }
    __syncthreads();

    // maxpool2 + relu -> h3p (2,32,2,4)
    for (int idx = tid; idx < 512; idx += 256) {
        int xw_ = idx & 3; int t = idx >> 2;
        int y   = t & 1;   t >>= 1;
        int ic  = t & 31;  int s = t >> 5;
        const _Float16* base = &s_h3[((s * 32 + ic) * 4 + 2 * y) * 8 + 2 * xw_];
        float m = fmaxf(fmaxf((float)base[0], (float)base[1]),
                        fmaxf((float)base[8], (float)base[9]));
        s_h3p[idx] = (_Float16)fmaxf(m, 0.f);
    }
    __syncthreads();

    // ================= conv4 (32->64, k2x4, VALID) : length-256 dot per (s,oc) =================
    if (tid < 128) {
        int s  = tid >> 6;
        int oc = tid & 63;
        float acc = b4g[oc];
        const float* wg = w4g + oc * 256;
        for (int k = 0; k < 256; ++k)
            acc += wg[k] * (float)s_h3p[s * 256 + k];
        s_h4[s * 64 + oc] = acc;
    }
    __syncthreads();

    // ================= fc (64->3) + softmax -> s_pred =================
    if (tid < 2) {
        int s = tid;
        float lg[3];
        for (int c = 0; c < 3; ++c) {
            float a = fcbg[c];
            for (int k = 0; k < 64; ++k) a += fcwg[c * 64 + k] * s_h4[s * 64 + k];
            lg[c] = a;
        }
        float mx = fmaxf(lg[0], fmaxf(lg[1], lg[2]));
        float e0 = expf(lg[0] - mx), e1 = expf(lg[1] - mx), e2 = expf(lg[2] - mx);
        float inv = 1.f / (e0 + e1 + e2);
        s_pred[s * 3 + 0] = e0 * inv;
        s_pred[s * 3 + 1] = e1 * inv;
        s_pred[s * 3 + 2] = e2 * inv;
    }
    __syncthreads();

    // ---- async LDS -> global store of the 6 preds (ASYNCcnt path) ----
    if (tid < 6) {
        int s = tid / 3, c = tid % 3;
        unsigned           lds_src = (unsigned)(size_t)&s_pred[tid];
        unsigned long long gaddr   =
            (unsigned long long)(size_t)(preds + ((size_t)(cb * 2 + s) * N_LEADS + l) * 3 + c);
        asm volatile("s_wait_dscnt 0" ::: "memory");
        asm volatile("global_store_async_from_lds_b32 %0, %1, off"
                     :: "v"(gaddr), "v"(lds_src) : "memory");
        wait_asynccnt0();
    }
}

// Cross-lead multiplicative fusion loss: out[b][c] = -sum_j (prod_{k!=j}(1-p_k))^(beta/(L-1)) * log p_j
__global__ void fusion_kernel(const float* __restrict__ preds, float* __restrict__ out)
{
    int t = blockIdx.x * blockDim.x + threadIdx.x;
    if (t >= N_BATCH * 3) return;
    int b = t / 3, c = t % 3;
    const float e = 2.0f / 60.0f;
    float prod = 1.f;
    for (int j = 0; j < N_LEADS; ++j)
        prod *= (1.f - preds[((size_t)b * N_LEADS + j) * 3 + c]);
    float loss = 0.f;
    for (int j = 0; j < N_LEADS; ++j) {
        float p  = preds[((size_t)b * N_LEADS + j) * 3 + c];
        float om = 1.f - p;
        float coeff = prod / om;
        loss += powf(coeff, e) * logf(p);
    }
    out[t] = -loss;
}

extern "C" void kernel_launch(void* const* d_in, const int* in_sizes, int n_in,
                              void* d_out, int out_size, void* d_ws, size_t ws_size,
                              hipStream_t stream)
{
    const float* x   = (const float*)d_in[0];
    const float* w1  = (const float*)d_in[1];
    const float* b1  = (const float*)d_in[2];
    const float* w2  = (const float*)d_in[3];
    const float* b2  = (const float*)d_in[4];
    const float* w3  = (const float*)d_in[5];
    const float* b3  = (const float*)d_in[6];
    const float* w4  = (const float*)d_in[7];
    const float* b4  = (const float*)d_in[8];
    const float* fcw = (const float*)d_in[9];
    const float* fcb = (const float*)d_in[10];

    float* preds = (float*)d_ws;            // 128*61*3 floats = 93,696 B
    dim3 grid(N_BATCH / 2, N_LEADS);        // 64 x 61 workgroups
    lead_cnn_kernel<<<grid, 256, 0, stream>>>(x, w1, b1, w2, b2, w3, b3, w4, b4,
                                              fcw, fcb, preds);
    fusion_kernel<<<dim3(3), 128, 0, stream>>>(preds, (float*)d_out);
}
